// Gru_15822659519158
// MI455X (gfx1250) — compile-verified, matrix operand order checked by reading, not run on hardware
//
#include <hip/hip_runtime.h>
#include <hip/hip_bf16.h>

// ---------------------------------------------------------------------------
// GRU scan, M=256, T=512, NX=256, U=256.
// Phase 1: grid GEMM xproj = x @ [wux|wrx|wcx] + bias, written in bf16 to d_ws
//          in the phase-2 register-tile layout: [t][mtile][ntile(48)][lane][8].
// Phase 2: 16 workgroups (16 batch rows each) run the 512-step scan WGP-local:
//          c state in VGPRs, A-operands staged in LDS in native WMMA A-layout,
//          wuc/wrc B-slices in VGPRs, wcc B-slices in LDS (320KB/WGP budget),
//          split workgroup barriers (s_barrier_signal/-wait) with global loads
//          pipelined into the barrier shadow.
// ---------------------------------------------------------------------------

typedef __attribute__((ext_vector_type(16))) __bf16 v16bf;
typedef __attribute__((ext_vector_type(8)))  __bf16 v8bf;
typedef __attribute__((ext_vector_type(8)))  float  v8f;

#define M_  256
#define T_  512
#define NX_ 256
#define U_  256

__device__ __forceinline__ __bf16 bf(float f) { return (__bf16)f; }

// 16-bit WMMA A/B operand element->K mapping (ISA 7.12.2):
// lane half 0 holds K {0..7, 16..23}, half 1 holds K {8..15, 24..31}
__device__ __forceinline__ int kmap(int half, int j) {
  return (j < 8) ? (half * 8 + j) : (16 + half * 8 + (j - 8));
}

__device__ __forceinline__ float sigmoidf_(float x) {
  return 1.0f / (1.0f + __expf(-x));
}
__device__ __forceinline__ float tanhf_(float x) {
  float e = __expf(2.0f * x);
  return (e - 1.0f) / (e + 1.0f);
}

// ---------------------------------------------------------------------------
// Phase 1. Block = 8 waves sharing one output column tile ct (of 48); each
// wave owns one row tile (16 rows of the (t,m)-flattened 131072-row matrix).
// B K-tiles staged once per block into LDS. Output stored as raw C-register
// tiles: xproj_v8[ ((t*16 + mtile)*48 + ct)*32 + lane ] = 8 bf16 (16B store).
// ---------------------------------------------------------------------------
__global__ __launch_bounds__(256)
void gru_xproj_kernel(const float* __restrict__ x,
                      const float* __restrict__ wux,
                      const float* __restrict__ wrx,
                      const float* __restrict__ wcx,
                      const float* __restrict__ bu,
                      const float* __restrict__ br,
                      const float* __restrict__ bc,
                      __hip_bfloat16* __restrict__ xproj_raw)
{
  v8bf* __restrict__ xpv = reinterpret_cast<v8bf*>(xproj_raw);
  __shared__ v16bf Bsh[8 * 32];                  // 8 K-tiles of B, 8KB

  const int lane = threadIdx.x & 31;
  const int wave = threadIdx.x >> 5;
  const int half = lane >> 4;
  const int rw   = lane & 15;

  const int ct = blockIdx.x % 48;                // column tile 0..47
  const int rg = blockIdx.x / 48;                // row-tile group 0..1023
  const int rowTile = rg * 8 + wave;             // 0..8191

  // column -> gate / weight
  const int n = ct * 16 + rw;                    // 0..767
  const int g = n >> 8;
  const int u = n & 255;
  const float* __restrict__ W  = (g == 0) ? wux : (g == 1) ? wrx : wcx;
  const float* __restrict__ Bb = (g == 0) ? bu  : (g == 1) ? br  : bc;

  // cooperative B stage: wave k loads K-tile k (shared by all 8 row tiles)
  {
    v16bf b;
    #pragma unroll
    for (int j = 0; j < 16; ++j) {
      const int k = wave * 32 + kmap(half, j);
      b[j] = bf(W[(size_t)k * U_ + u]);
    }
    Bsh[wave * 32 + lane] = b;
  }
  __syncthreads();

  // row tile -> (t, mtile); all 16 rows of a tile share t (256 % 16 == 0)
  const int t  = rowTile >> 4;
  const int mt = rowTile & 15;
  const int m  = mt * 16 + rw;                   // A-row for this lane
  const float* __restrict__ xrow = x + ((size_t)m * T_ + t) * NX_;

  const float bv = Bb[u];
  v8f acc = { bv, bv, bv, bv, bv, bv, bv, bv };

  #pragma unroll
  for (int kt = 0; kt < 8; ++kt) {
    const int k0 = kt * 32;
    v16bf a;
    const float* s0 = xrow + k0 + half * 8;      // two contiguous 8-float runs
    const float* s1 = s0 + 16;
    #pragma unroll
    for (int j = 0; j < 8; ++j) { a[j] = bf(s0[j]); a[j + 8] = bf(s1[j]); }
    acc = __builtin_amdgcn_wmma_f32_16x16x32_bf16(false, a, false, Bsh[kt * 32 + lane],
                                                  (short)0, acc, false, false);
  }

  // store the 16x16 tile in raw C-register layout: one 16B store per lane
  v8bf o;
  #pragma unroll
  for (int r = 0; r < 8; ++r) o[r] = bf(acc[r]);
  xpv[(((size_t)t * 16 + mt) * 48 + ct) * 32 + lane] = o;
}

// ---------------------------------------------------------------------------
// Phase 2: recurrent scan. 16 blocks x 512 threads (16 waves).
// Block mb owns batch rows [16mb,16mb+16); wave w owns hidden cols [16w,16w+16).
// Dynamic LDS: cA (8KB, A-layout of c), grA (8KB, A-layout of g_r*c),
//              wccL (128KB, per-wave B-slices of wcc). Total 144KB.
// ---------------------------------------------------------------------------
__global__ __launch_bounds__(512, 1)
void gru_scan_kernel(const float* __restrict__ a0,
                     const float* __restrict__ wuc,
                     const float* __restrict__ wrc,
                     const float* __restrict__ wcc,
                     const __hip_bfloat16* __restrict__ xproj_raw,
                     float* __restrict__ out)
{
  const v8bf* __restrict__ xpv = reinterpret_cast<const v8bf*>(xproj_raw);

  extern __shared__ __align__(32) char smem[];
  v16bf* __restrict__ cA   = (v16bf*)(smem);              //  8 KB
  v16bf* __restrict__ grA  = (v16bf*)(smem + 8 * 1024);   //  8 KB
  v16bf* __restrict__ wccL = (v16bf*)(smem + 16 * 1024);  // 128 KB
  __bf16* __restrict__ cA_s  = (__bf16*)cA;
  __bf16* __restrict__ grA_s = (__bf16*)grA;

  const int lane = threadIdx.x & 31;
  const int w    = threadIdx.x >> 5;          // wave 0..15 = hidden-column tile
  const int half = lane >> 4;
  const int rw   = lane & 15;
  const int mb   = blockIdx.x;                // batch-row tile 0..15
  const int n    = w * 16 + rw;               // hidden column owned by this lane

  // --- one-time: load recurrent weight B-slices ---
  v16bf Bu[8], Br[8];
  #pragma unroll
  for (int kt = 0; kt < 8; ++kt) {
    v16bf tu, tr, tc;
    #pragma unroll
    for (int j = 0; j < 16; ++j) {
      const int k = kt * 32 + kmap(half, j);
      tu[j] = bf(wuc[(size_t)k * U_ + n]);
      tr[j] = bf(wrc[(size_t)k * U_ + n]);
      tc[j] = bf(wcc[(size_t)k * U_ + n]);
    }
    Bu[kt] = tu;
    Br[kt] = tr;
    wccL[(w * 8 + kt) * 32 + lane] = tc;      // wcc slice -> LDS (VGPR relief)
  }

  // --- one-time: initial state into C-layout registers ---
  v8f c;
  #pragma unroll
  for (int r = 0; r < 8; ++r)
    c[r] = a0[(size_t)(mb * 16 + r + half * 8) * U_ + n];

  // A-layout write slot for (row, K=n): invert kmap once per lane.
  const int kt_n = n >> 5;
  const int kp   = n & 31;
  int halfp, jp;
  if (kp < 16) { halfp = kp >> 3; jp = kp & 7; }
  else         { const int kk = kp - 16; halfp = kk >> 3; jp = (kk & 7) + 8; }
  const int slotBase = kt_n * 512 + halfp * 256 + jp;   // + row*16 per element

  #pragma unroll
  for (int r = 0; r < 8; ++r)
    cA_s[slotBase + (r + half * 8) * 16] = bf(c[r]);

  __syncthreads();

  // preload step-0 u/r x-projection tiles (one b128 each)
  v8bf xu = xpv[((size_t)mb * 48 + w) * 32 + lane];
  v8bf xr = xpv[((size_t)mb * 48 + 16 + w) * 32 + lane];

  for (int t = 0; t < T_; ++t) {
    const size_t tb = ((size_t)t * 16 + mb) * 48;

    v8f au, ar;
    #pragma unroll
    for (int r = 0; r < 8; ++r) { au[r] = (float)xu[r]; ar[r] = (float)xr[r]; }

    // au += c @ wuc ; ar += c @ wrc
    #pragma unroll
    for (int kt = 0; kt < 8; ++kt) {
      const v16bf a = cA[kt * 32 + lane];
      au = __builtin_amdgcn_wmma_f32_16x16x32_bf16(false, a, false, Bu[kt],
                                                   (short)0, au, false, false);
      ar = __builtin_amdgcn_wmma_f32_16x16x32_bf16(false, a, false, Br[kt],
                                                   (short)0, ar, false, false);
    }

    v8f gu, grc;
    #pragma unroll
    for (int r = 0; r < 8; ++r) {
      gu[r]  = sigmoidf_(au[r]);
      grc[r] = sigmoidf_(ar[r]) * c[r];
    }

    // publish g_r*c, then split barrier with the candidate-gate load in its shadow
    #pragma unroll
    for (int r = 0; r < 8; ++r)
      grA_s[slotBase + (r + half * 8) * 16] = bf(grc[r]);
    asm volatile("s_wait_dscnt 0x0\n\ts_barrier_signal -1" ::: "memory");
    const v8bf xc = xpv[(tb + 32 + w) * 32 + lane];     // global load in barrier shadow
    asm volatile("s_barrier_wait -1" ::: "memory");

    // candidate: tanh((g_r*c) @ wcc + x_t @ wcx + bc)
    v8f ac;
    #pragma unroll
    for (int r = 0; r < 8; ++r) ac[r] = (float)xc[r];
    #pragma unroll
    for (int kt = 0; kt < 8; ++kt) {
      const v16bf a   = grA[kt * 32 + lane];
      const v16bf bc_ = wccL[(w * 8 + kt) * 32 + lane];
      ac = __builtin_amdgcn_wmma_f32_16x16x32_bf16(false, a, false, bc_,
                                                   (short)0, ac, false, false);
    }

    // state update (f32 registers)
    #pragma unroll
    for (int r = 0; r < 8; ++r) {
      const float cand = tanhf_(ac[r]);
      c[r] = c[r] + gu[r] * (cand - c[r]);
    }

    // republish c; next step's u/r tiles load in the barrier shadow
    #pragma unroll
    for (int r = 0; r < 8; ++r)
      cA_s[slotBase + (r + half * 8) * 16] = bf(c[r]);
    asm volatile("s_wait_dscnt 0x0\n\ts_barrier_signal -1" ::: "memory");
    const size_t tn = ((size_t)((t + 1 < T_) ? t + 1 : t) * 16 + mb) * 48;
    xu = xpv[(tn + w) * 32 + lane];
    xr = xpv[(tn + 16 + w) * 32 + lane];
    asm volatile("s_barrier_wait -1" ::: "memory");
  }

  // final hidden state, f32
  #pragma unroll
  for (int r = 0; r < 8; ++r)
    out[(size_t)(mb * 16 + r + half * 8) * U_ + n] = c[r];
}

// ---------------------------------------------------------------------------
extern "C" void kernel_launch(void* const* d_in, const int* in_sizes, int n_in,
                              void* d_out, int out_size, void* d_ws, size_t ws_size,
                              hipStream_t stream) {
  const float* x   = (const float*)d_in[0];
  const float* a0  = (const float*)d_in[1];
  const float* wcx = (const float*)d_in[2];
  const float* wcc = (const float*)d_in[3];
  const float* bc  = (const float*)d_in[4];
  const float* wux = (const float*)d_in[5];
  const float* wuc = (const float*)d_in[6];
  const float* bu  = (const float*)d_in[7];
  const float* wrx = (const float*)d_in[8];
  const float* wrc = (const float*)d_in[9];
  const float* br  = (const float*)d_in[10];

  float* out = (float*)d_out;
  __hip_bfloat16* xproj = (__hip_bfloat16*)d_ws;   // T*M*3U bf16 = 201 MB, tiled

  // Phase 1: 48 column tiles x 1024 row-tile groups (8 row tiles per block)
  gru_xproj_kernel<<<49152, 256, 0, stream>>>(x, wux, wrx, wcx, bu, br, bc, xproj);

  // Phase 2: 16 row-blocks, 16 waves each, 144 KB dynamic LDS
  gru_scan_kernel<<<16, 512, 147456, stream>>>(a0, wuc, wrc, wcc, xproj, out);
}
